// Model_39676907881182
// MI455X (gfx1250) — matrix operand order repaired and checked by reading, not compiled, not run on hardware
//
#include <hip/hip_runtime.h>

#define NN   1024   // padded spatial size
#define HPAD 1022   // unpadded H
#define CIN  3
#define BR   32     // Q rows per block
#define BC   32     // keys per inner iteration
#define WCH  256    // output columns per block
#define NTHREADS 128

#if defined(__HIP_DEVICE_COMPILE__) && __has_builtin(__builtin_amdgcn_tensor_load_to_lds) && __has_builtin(__builtin_amdgcn_s_wait_tensorcnt)
#define HAVE_TDM 1
#else
#define HAVE_TDM 0
#endif

typedef __attribute__((ext_vector_type(16))) _Float16     v16h;
typedef __attribute__((ext_vector_type(8)))  _Float16     v8h;
typedef __attribute__((ext_vector_type(8)))  float        v8f;
typedef __attribute__((ext_vector_type(4)))  unsigned int v4u;
typedef __attribute__((ext_vector_type(8)))  int          v8i;
typedef __attribute__((ext_vector_type(4)))  int          v4i;

// Build a 16-element f16 WMMA fragment from two contiguous 8-half LDS chunks.
static __device__ __forceinline__ v16h make_frag(const _Float16* lo_p, const _Float16* hi_p) {
  v8h lo = *(const v8h*)lo_p;
  v8h hi = *(const v8h*)hi_p;
  v16h r;
#pragma unroll
  for (int i = 0; i < 8; ++i) { r[i] = lo[i]; r[i + 8] = hi[i]; }
  return r;
}

__global__ __launch_bounds__(NTHREADS, 1)
void fused_conv_attn(const float* __restrict__ x1, const float* __restrict__ x2,
                     const float* __restrict__ Wq, const float* __restrict__ bq,
                     const float* __restrict__ Wk, const float* __restrict__ bk,
                     const float* __restrict__ Wv, const float* __restrict__ bv,
                     const unsigned char* __restrict__ dmask,
                     float* __restrict__ out)
{
  // CDNA5: 320 KB LDS per WGP -> keep full Q and K row-tiles resident (~151 KB total).
  __shared__ __align__(16) _Float16 Qs[BR][NN];     // 64 KB  Q tile (f16), full width, built once
  __shared__ __align__(16) _Float16 Ks[BC][NN];     // 64 KB  K tile (f16), full width, per key-block
  __shared__ __align__(16) _Float16 VTs[WCH][BC];   // 16 KB  V tile transposed (w-major)
  __shared__ __align__(16) _Float16 Ps[BR][BC];     //  2 KB  masked exp() probabilities
  __shared__ float Ss[BR][BC];                      //  4 KB  raw scores
  __shared__ __align__(8) unsigned char Ms[BR][BC]; //  1 KB  dropout mask tile (TDM destination)
  __shared__ float alpha_s[BR];
  __shared__ float l_s[BR];

  const int tid  = threadIdx.x;
  const int lane = tid & 31;
  const int wave = tid >> 5;
  const int wm   = wave >> 1;              // 16-row half of the 32-row tile
  const int wn   = wave & 1;               // S col half / O 128-col half
  const int lm   = lane & 15;
  const int ahalf = (lane < 16) ? 0 : 8;   // A-fragment K offset (ISA 16-bit A layout)
  const int bbase = (lane < 16) ? 0 : 8;   // B-fragment K offset (same table order as A)

  const int pair = blockIdx.z;             // b * 8 + c_out
  const int b    = pair >> 3;
  const int c    = pair & 7;
  const int r0   = blockIdx.x * BR;        // global Q-row base
  const int w0   = blockIdx.y * WCH;       // global output-col base

  const size_t CH2 = (size_t)HPAD * HPAD;
  const float* x1b = x1 + (size_t)b * CIN * CH2;
  const float* x2b = x2 + (size_t)b * CIN * CH2;

  const float wq0 = Wq[c*CIN+0], wq1 = Wq[c*CIN+1], wq2 = Wq[c*CIN+2], bqv = bq[c];
  const float wk0 = Wk[c*CIN+0], wk1 = Wk[c*CIN+1], wk2 = Wk[c*CIN+2], bkv = bk[c];
  const float wv0 = Wv[c*CIN+0], wv1 = Wv[c*CIN+1], wv2 = Wv[c*CIN+2], bvv = bv[c];

  // ---- Phase A: build full Q tile once (fused conv1x1 + zero pad, f32 -> f16) ----
  for (int idx = tid; idx < BR * NN; idx += NTHREADS) {
    const int hl = idx >> 10, w = idx & (NN - 1);
    const int h = r0 + hl;
    float q = bqv;
    if (h >= 1 && h <= HPAD && w >= 1 && w <= HPAD) {
      const float* p = x1b + (size_t)(h - 1) * HPAD + (w - 1);
      q += wq0 * p[0] + wq1 * p[CH2] + wq2 * p[2 * CH2];
    }
    Qs[hl][w] = (_Float16)q;
  }

  const v8f vzero = {0.f,0.f,0.f,0.f,0.f,0.f,0.f,0.f};
  v8f oacc[8];                             // 16 rows x 128 cols of O per wave (f32)
#pragma unroll
  for (int t = 0; t < 8; ++t) oacc[t] = vzero;

  float m_run = -3.0e38f;                  // per-row running max (threads 0..31 own row=tid)
  float l_run = 0.0f;                      // per-row running exp-sum (unmasked)

  for (int jb = 0; jb < NN / BC; ++jb) {
    const int g0 = jb * BC;

#if HAVE_TDM
    // ---- issue TDM load of the 32x32 dropout-mask tile; overlaps K/V staging + S GEMM ----
    if (wave == 0) {
      const unsigned lds_off = (unsigned)(unsigned long long)(uintptr_t)(&Ms[0][0]);
      const unsigned long long ga =
          (unsigned long long)(uintptr_t)(dmask + ((size_t)pair * NN + r0) * NN + g0);
      v4u g0d;
      g0d[0] = 1u;                                           // count=1, user descriptor
      g0d[1] = lds_off;                                      // lds_addr
      g0d[2] = (unsigned)(ga & 0xffffffffull);               // global_addr[31:0]
      g0d[3] = (unsigned)((ga >> 32) & 0x01ffffffull) | (2u << 30); // addr[56:32] | type=2
      v8i g1d;
      g1d[0] = 0;                                            // wg_mask=0, data_size=1B
      g1d[1] = (int)((NN & 0xffff) << 16);                   // tensor_dim0[15:0]
      g1d[2] = (int)(((NN >> 16) & 0xffff) | ((NN & 0xffff) << 16)); // dim0 hi | dim1 lo
      g1d[3] = (int)(((NN >> 16) & 0xffff) | (BC << 16));    // dim1 hi | tile_dim0=32
      g1d[4] = (int)BR;                                      // tile_dim1=32, tile_dim2=0
      g1d[5] = (int)NN;                                      // tensor_dim0_stride[31:0]
      g1d[6] = 0;                                            // stride0 hi | stride1 lo
      g1d[7] = 0;
      const v4i z4 = {0, 0, 0, 0};
#if defined(__clang_major__) && (__clang_major__ >= 23)
      const v8i z8 = {0, 0, 0, 0, 0, 0, 0, 0};
      __builtin_amdgcn_tensor_load_to_lds(g0d, g1d, z4, z4, z8, 0);
#else
      __builtin_amdgcn_tensor_load_to_lds(g0d, g1d, z4, z4, 0);
#endif
    }
#endif

    // ---- stage full K rows (+V in our output window) from x2, fused conv1x1 ----
    for (int idx = tid; idx < BC * NN; idx += NTHREADS) {
      const int gl = idx >> 10, w = idx & (NN - 1);
      const int g = g0 + gl;
      float kq = bkv, vq = bvv;
      if (g >= 1 && g <= HPAD && w >= 1 && w <= HPAD) {
        const float* p = x2b + (size_t)(g - 1) * HPAD + (w - 1);
        const float a0 = p[0], a1 = p[CH2], a2 = p[2 * CH2];
        kq += wk0 * a0 + wk1 * a1 + wk2 * a2;
        vq += wv0 * a0 + wv1 * a1 + wv2 * a2;
        if (g + BC <= HPAD) __builtin_prefetch(p + (size_t)BC * HPAD, 0, 0);
      }
      Ks[gl][w] = (_Float16)kq;
      const unsigned wd = (unsigned)(w - w0);
      if (wd < WCH) VTs[wd][gl] = (_Float16)vq;
    }
    __syncthreads();

    // ---- S = Q x K^T over full 1024-wide inner dim (v_wmma_f32_16x16x32_f16) ----
    v8f sacc = vzero;
    const _Float16* qrow = &Qs[wm * 16 + lm][0];
    const _Float16* krow = &Ks[wn * 16 + lm][0];
#pragma unroll 8
    for (int kk = 0; kk < NN; kk += 32) {
      v16h A  = make_frag(qrow + kk + ahalf, qrow + kk + ahalf + 16);
      v16h Bf = make_frag(krow + kk + bbase, krow + kk + bbase + 16);
      sacc = __builtin_amdgcn_wmma_f32_16x16x32_f16(false, A, false, Bf,
                                                    (short)0, sacc, false, false);
    }

    // ---- spill S tile (apply 1/10 scale) ----
#pragma unroll
    for (int r = 0; r < 8; ++r)
      Ss[wm * 16 + ahalf + r][wn * 16 + lm] = sacc[r] * 0.1f;
#if HAVE_TDM
    if (wave == 0) __builtin_amdgcn_s_wait_tensorcnt(0);
#endif
    __syncthreads();

    // ---- online softmax + dropout -> P (wave 0; thread `row` owns row stats) ----
    if (tid < BR) {
      const int row = tid;
      float mc = Ss[row][0];
#pragma unroll
      for (int cc = 1; cc < BC; ++cc) mc = fmaxf(mc, Ss[row][cc]);
      const float mnew = fmaxf(m_run, mc);
      const float al = __expf(m_run - mnew);
#if !HAVE_TDM
      const unsigned char* mrow = dmask + ((size_t)pair * NN + (r0 + row)) * NN + g0;
#endif
      float s = 0.f;
#pragma unroll
      for (int cc = 0; cc < BC; ++cc) {
        const float p = __expf(Ss[row][cc] - mnew);
        s += p;                                   // denominator is UNmasked
#if HAVE_TDM
        const unsigned char mb = Ms[row][cc];
#else
        const unsigned char mb = mrow[cc];
#endif
        Ps[row][cc] = (_Float16)(mb ? 2.0f * p : 0.0f);  // numerator masked * 1/(1-p)
      }
      l_run = l_run * al + s;
      m_run = mnew;
      alpha_s[row] = al;
    }
    __syncthreads();

    // ---- rescale O by alpha, then O += P x V (WMMA) ----
    float af[8];
#pragma unroll
    for (int r = 0; r < 8; ++r) af[r] = alpha_s[wm * 16 + ahalf + r];
#pragma unroll
    for (int t = 0; t < 8; ++t)
#pragma unroll
      for (int r = 0; r < 8; ++r) oacc[t][r] *= af[r];

    const _Float16* prow = &Ps[wm * 16 + lm][0];
    const v16h Pa = make_frag(prow + ahalf, prow + ahalf + 16);
#pragma unroll
    for (int t = 0; t < 8; ++t) {
      const _Float16* vcol = &VTs[wn * 128 + t * 16 + lm][bbase];
      v16h Bv = make_frag(vcol, vcol + 16);
      oacc[t] = __builtin_amdgcn_wmma_f32_16x16x32_f16(false, Pa, false, Bv,
                                                       (short)0, oacc[t], false, false);
    }
    __syncthreads();
  }

  // ---- finalize: O / l, store f32 ----
  if (tid < BR) l_s[tid] = l_run;
  __syncthreads();
  float linv[8];
#pragma unroll
  for (int r = 0; r < 8; ++r) linv[r] = 1.0f / l_s[wm * 16 + ahalf + r];
  float* ob = out + (size_t)pair * NN * NN;
#pragma unroll
  for (int t = 0; t < 8; ++t) {
    const int wcol = w0 + wn * 128 + t * 16 + lm;
#pragma unroll
    for (int r = 0; r < 8; ++r) {
      const int row = r0 + wm * 16 + ahalf + r;
      ob[(size_t)row * NN + wcol] = oacc[t][r] * linv[r];
    }
  }
}

extern "C" void kernel_launch(void* const* d_in, const int* in_sizes, int n_in,
                              void* d_out, int out_size, void* d_ws, size_t ws_size,
                              hipStream_t stream) {
  (void)in_sizes; (void)n_in; (void)out_size; (void)d_ws; (void)ws_size;
  const float* x1 = (const float*)d_in[0];
  const float* x2 = (const float*)d_in[1];
  const float* Wq = (const float*)d_in[2];
  const float* bq = (const float*)d_in[3];
  const float* Wk = (const float*)d_in[4];
  const float* bk = (const float*)d_in[5];
  const float* Wv = (const float*)d_in[6];
  const float* bv = (const float*)d_in[7];
  const unsigned char* dm = (const unsigned char*)d_in[8];
  float* out = (float*)d_out;

  dim3 grid(NN / BR, NN / WCH, 8 * 8);   // (row blocks, w-chunks, B*Cout)
  dim3 block(NTHREADS);
  hipLaunchKernelGGL(fused_conv_attn, grid, block, 0, stream,
                     x1, x2, Wq, bq, Wk, bk, Wv, bv, dm, out);
}
